// fstgcn_61469571940997
// MI455X (gfx1250) — matrix operand order repaired and checked by reading, not compiled, not run on hardware
//
#include <hip/hip_runtime.h>
#include <hip/hip_bf16.h>

typedef __bf16 bf16_t;
typedef __attribute__((ext_vector_type(16))) __bf16 v16bf;
typedef __attribute__((ext_vector_type(8)))  float  v8f;

#define WMMA_BF16(a, b, c) \
  __builtin_amdgcn_wmma_f32_16x16x32_bf16(false, (a), false, (b), (short)0, (c), false, false)

// ---------------- constants ----------------
// B=32 T=12 N=1024 C=64  BC=2048  NW=9 NL=3  TN=T*N=12288
#define CB   32
#define CT   12
#define CN   1024
#define CC   64
#define CBC  2048
#define CNW  9
#define CTN  12288

// ---------------- ISA-layout tile loaders ----------------
// A-matrix 16x32 bf16 per wave: lane holds row (lane&15); element i<8 -> K = 8*(lane>=16)+i,
// element i>=8 -> K = 16 + 8*(lane>=16) + (i-8). Two contiguous 16B chunks.
__device__ __forceinline__ v16bf loadA16(const bf16_t* rowbase, int k0, int lane) {
  const bf16_t* p = rowbase + k0 + ((lane >> 4) << 3);
  union { uint4 u[2]; v16bf v; } t;
  t.u[0] = *reinterpret_cast<const uint4*>(p);
  t.u[1] = *reinterpret_cast<const uint4*>(p + 16);
  return t.v;
}
// B-matrix 32x16 bf16: lane holds column (lane&15); element i<8 -> K = 8*(lane>=16)+i,
// element i>=8 -> K = 16 + 8*(lane>=16) + (i-8). Two 16B loads. rowbase already points at row (column of B).
__device__ __forceinline__ v16bf loadB16(const bf16_t* rowbase, int k0, int lane) {
  const bf16_t* p = rowbase + k0 + ((lane >> 4) << 3);
  union { uint4 u[2]; v16bf v; } t;
  t.u[0] = *reinterpret_cast<const uint4*>(p);
  t.u[1] = *reinterpret_cast<const uint4*>(p + 16);
  return t.v;
}

__device__ __forceinline__ float sigmoidf_(float x) {
  return 1.0f / (1.0f + __expf(-x));
}

// ---------------- prep kernels ----------------
// xrT[b*64+c][t*1024+n] = bf16(data[b,t,n,c] + temb[t,c] + semb[n,c])
__global__ void prep_xr(const float* __restrict__ data, const float* __restrict__ temb,
                        const float* __restrict__ semb, bf16_t* __restrict__ xrT) {
  __shared__ float tile[64][65];
  int bid = blockIdx.x;                 // 32*12*16 blocks
  int nblk = bid & 15;
  int rem  = bid >> 4;
  int t = rem % CT;
  int b = rem / CT;
  int n0 = nblk * 64;
#pragma unroll
  for (int it = 0; it < 16; ++it) {
    int idx = it * 256 + threadIdx.x;   // 64n x 64c tile
    int nn = idx >> 6, c = idx & 63;
    tile[nn][c] = data[(((size_t)b * CT + t) * CN + n0 + nn) * CC + c]
                + temb[t * CC + c] + semb[(n0 + nn) * CC + c];
  }
  __syncthreads();
  int c   = threadIdx.x >> 2;
  int nn0 = (threadIdx.x & 3) << 4;
  bf16_t* dst = xrT + (size_t)(b * CC + c) * CTN + t * CN + n0 + nn0;
#pragma unroll
  for (int q = 0; q < 16; ++q) dst[q] = (bf16_t)tile[nn0 + q][c];
}

__global__ void prep_adj(const float* __restrict__ adj, bf16_t* __restrict__ adjb) {
  size_t i = (size_t)blockIdx.x * blockDim.x + threadIdx.x;  // 1024*4096 elems
  adjb[i] = (bf16_t)adj[i];
}

// WcT[oc][k*64+ic] = conv1_w[oc, ic, 0, k]   (conv1_w is (128,64,1,4))
__global__ void prep_wconv(const float* __restrict__ w1, bf16_t* __restrict__ WcT) {
  int i = blockIdx.x * blockDim.x + threadIdx.x;  // 32768
  int oc = i >> 8, kk = i & 255;
  int k = kk >> 6, ic = kk & 63;
  WcT[i] = (bf16_t)w1[((oc * 64) + ic) * 4 + k];
}

// gwT[w*3+j][d][c] = gcn_w[w, j, c, d]   (gcn_w is (9,3,64,128))
__global__ void prep_gcnw(const float* __restrict__ gw, bf16_t* __restrict__ gwT) {
  int i = blockIdx.x * blockDim.x + threadIdx.x;  // 27*8192 = 221184
  int wj = i >> 13, rem = i & 8191;
  int d = rem >> 6, c = rem & 63;
  gwT[i] = (bf16_t)gw[(size_t)wj * 8192 + c * 128 + d];
}

// ---------------- conv (temporal) + GLU, writes data_res straight into d_out --------
// per (b, t'): out[n, c] = sigmoid(L[n,c]) * R[n,c], with [L|R] = data-window GEMM (M=1024,K=256,N=128)
__global__ void conv_glu(const float* __restrict__ data, const bf16_t* __restrict__ WcT,
                         const float* __restrict__ cbias, float* __restrict__ out) {
  int lane = threadIdx.x & 31;
  int gw   = blockIdx.x * (blockDim.x >> 5) + (threadIdx.x >> 5); // 18432 waves
  int mt = gw & 63;
  int bt = gw >> 6;             // 0..287
  int b  = bt / CNW;
  int tp = bt - b * CNW;
  int n0 = mt * 16;
  int lh = lane >> 4, lj = lane & 15;
  int n = n0 + lj;

  v8f acc[8];
#pragma unroll
  for (int j = 0; j < 8; ++j) acc[j] = (v8f){0.f,0.f,0.f,0.f,0.f,0.f,0.f,0.f};

  // A (16x32 bf16) built on the fly from f32 data
  auto loadA = [&](int k0) -> v16bf {
    int base0 = k0 + (lh << 3);
    int base1 = base0 + 16;
    const float* p0 = data + (((size_t)b * CT + tp + (base0 >> 6)) * CN + n) * CC + (base0 & 63);
    const float* p1 = data + (((size_t)b * CT + tp + (base1 >> 6)) * CN + n) * CC + (base1 & 63);
    float4 f0 = *reinterpret_cast<const float4*>(p0);
    float4 f1 = *reinterpret_cast<const float4*>(p0 + 4);
    float4 f2 = *reinterpret_cast<const float4*>(p1);
    float4 f3 = *reinterpret_cast<const float4*>(p1 + 4);
    v16bf a;
    a[0]=(__bf16)f0.x; a[1]=(__bf16)f0.y; a[2]=(__bf16)f0.z; a[3]=(__bf16)f0.w;
    a[4]=(__bf16)f1.x; a[5]=(__bf16)f1.y; a[6]=(__bf16)f1.z; a[7]=(__bf16)f1.w;
    a[8]=(__bf16)f2.x; a[9]=(__bf16)f2.y; a[10]=(__bf16)f2.z; a[11]=(__bf16)f2.w;
    a[12]=(__bf16)f3.x; a[13]=(__bf16)f3.y; a[14]=(__bf16)f3.z; a[15]=(__bf16)f3.w;
    return a;
  };

  // software pipeline: double-buffered operand stages over K=256 (8 steps -> 4 double steps)
  v16bf a0 = loadA(0);
  v16bf b0[8];
#pragma unroll
  for (int j = 0; j < 8; ++j) b0[j] = loadB16(WcT + (size_t)(j * 16 + lj) * 256, 0, lane);

#pragma unroll
  for (int k0 = 0; k0 < 256; k0 += 64) {
    v16bf a1 = loadA(k0 + 32);
    v16bf b1[8];
#pragma unroll
    for (int j = 0; j < 8; ++j) b1[j] = loadB16(WcT + (size_t)(j * 16 + lj) * 256, k0 + 32, lane);
#pragma unroll
    for (int j = 0; j < 8; ++j) acc[j] = WMMA_BF16(a0, b0[j], acc[j]);
    if (k0 + 64 < 256) {
      a0 = loadA(k0 + 64);
#pragma unroll
      for (int j = 0; j < 8; ++j) b0[j] = loadB16(WcT + (size_t)(j * 16 + lj) * 256, k0 + 64, lane);
    }
#pragma unroll
    for (int j = 0; j < 8; ++j) acc[j] = WMMA_BF16(a1, b1[j], acc[j]);
  }

  // GLU epilogue: data_res = sigmoid(left(c)) * right(c+64)
#pragma unroll
  for (int j = 0; j < 4; ++j) {
    float bl = cbias[j * 16 + lj];
    float br = cbias[64 + j * 16 + lj];
#pragma unroll
    for (int r = 0; r < 8; ++r) {
      float zl = acc[j][r] + bl;
      float zr = acc[j + 4][r] + br;
      int nn = n0 + r + (lh << 3);
      out[(((size_t)b * CNW + tp) * CN + nn) * CC + j * 16 + lj] = sigmoidf_(zl) * zr;
    }
  }
}

// ---------------- big GEMM: Y(1024x2048) = A(1024xK) * Bt^T, bf16 WMMA, f32 acc -------
// mode 0: acc from zero, store f32 to Yf32 (wing partial)
// mode 1: acc init from Cinit (f32), store bf16 to Ybf (per-layer result)
__global__ void gemm_big(const bf16_t* __restrict__ A, int lda,
                         const bf16_t* __restrict__ Bt, int ldb,
                         const float* __restrict__ Cinit,
                         float* __restrict__ Yf32, bf16_t* __restrict__ Ybf,
                         int Ksz, int mode) {
  int lane = threadIdx.x & 31;
  int gw   = blockIdx.x * (blockDim.x >> 5) + (threadIdx.x >> 5); // 512 waves
  int tileM = gw & 15;
  int tileN = gw >> 4;
  int r0 = tileM * 64, c0 = tileN * 64;
  int lh = lane >> 4, lj = lane & 15;

  v8f acc[4][4];
  if (mode) {
#pragma unroll
    for (int i = 0; i < 4; ++i)
#pragma unroll
      for (int j = 0; j < 4; ++j)
#pragma unroll
        for (int r = 0; r < 8; ++r)
          acc[i][j][r] = Cinit[(size_t)(r0 + i * 16 + r + (lh << 3)) * CBC + c0 + j * 16 + lj];
  } else {
#pragma unroll
    for (int i = 0; i < 4; ++i)
#pragma unroll
      for (int j = 0; j < 4; ++j)
        acc[i][j] = (v8f){0.f,0.f,0.f,0.f,0.f,0.f,0.f,0.f};
  }

  v16bf a0[4], b0[4], a1[4], b1[4];
#pragma unroll
  for (int i = 0; i < 4; ++i) a0[i] = loadA16(A + (size_t)(r0 + i * 16 + lj) * lda, 0, lane);
#pragma unroll
  for (int j = 0; j < 4; ++j) b0[j] = loadB16(Bt + (size_t)(c0 + j * 16 + lj) * ldb, 0, lane);

  // Ksz is a multiple of 64 (3072 or 1024): 2x-unrolled, double-buffered pipeline
  for (int k0 = 0; k0 < Ksz; k0 += 64) {
#pragma unroll
    for (int i = 0; i < 4; ++i) a1[i] = loadA16(A + (size_t)(r0 + i * 16 + lj) * lda, k0 + 32, lane);
#pragma unroll
    for (int j = 0; j < 4; ++j) b1[j] = loadB16(Bt + (size_t)(c0 + j * 16 + lj) * ldb, k0 + 32, lane);
#pragma unroll
    for (int i = 0; i < 4; ++i)
#pragma unroll
      for (int j = 0; j < 4; ++j)
        acc[i][j] = WMMA_BF16(a0[i], b0[j], acc[i][j]);
    if (k0 + 64 < Ksz) {
#pragma unroll
      for (int i = 0; i < 4; ++i) a0[i] = loadA16(A + (size_t)(r0 + i * 16 + lj) * lda, k0 + 64, lane);
#pragma unroll
      for (int j = 0; j < 4; ++j) b0[j] = loadB16(Bt + (size_t)(c0 + j * 16 + lj) * ldb, k0 + 64, lane);
    }
#pragma unroll
    for (int i = 0; i < 4; ++i)
#pragma unroll
      for (int j = 0; j < 4; ++j)
        acc[i][j] = WMMA_BF16(a1[i], b1[j], acc[i][j]);
  }

  if (mode == 0) {
#pragma unroll
    for (int i = 0; i < 4; ++i)
#pragma unroll
      for (int j = 0; j < 4; ++j)
#pragma unroll
        for (int r = 0; r < 8; ++r)
          Yf32[(size_t)(r0 + i * 16 + r + (lh << 3)) * CBC + c0 + j * 16 + lj] = acc[i][j][r];
  } else {
#pragma unroll
    for (int i = 0; i < 4; ++i)
#pragma unroll
      for (int j = 0; j < 4; ++j)
#pragma unroll
        for (int r = 0; r < 8; ++r)
          Ybf[(size_t)(r0 + i * 16 + r + (lh << 3)) * CBC + c0 + j * 16 + lj] =
              (bf16_t)acc[i][j][r];
  }
}

// ---------------- small GEMM Z = Y @ W_j + b_j, fused GLU + max + curT transpose -------
// M = 32768 rows (n*32+b), K = 64, N = 128. Wave does 16 rows x 128 cols.
__global__ void gemm_glu(const bf16_t* __restrict__ Ybf, const bf16_t* __restrict__ Wt,
                         const float* __restrict__ bias, bf16_t* __restrict__ curT,
                         float* __restrict__ outmax, int layer) {
  int lane = threadIdx.x & 31;
  int gw   = blockIdx.x * (blockDim.x >> 5) + (threadIdx.x >> 5); // 2048 waves
  int r0 = gw * 16;
  int lh = lane >> 4, lj = lane & 15;

  v8f acc[8];
#pragma unroll
  for (int j = 0; j < 8; ++j) acc[j] = (v8f){0.f,0.f,0.f,0.f,0.f,0.f,0.f,0.f};

  // K=64: preload both K-stages entirely, then back-to-back WMMAs
  v16bf a0 = loadA16(Ybf + (size_t)(r0 + lj) * 64, 0, lane);
  v16bf a1 = loadA16(Ybf + (size_t)(r0 + lj) * 64, 32, lane);
  v16bf bb0[8], bb1[8];
#pragma unroll
  for (int j = 0; j < 8; ++j) bb0[j] = loadB16(Wt + (size_t)(j * 16 + lj) * 64, 0, lane);
#pragma unroll
  for (int j = 0; j < 8; ++j) bb1[j] = loadB16(Wt + (size_t)(j * 16 + lj) * 64, 32, lane);
#pragma unroll
  for (int j = 0; j < 8; ++j) acc[j] = WMMA_BF16(a0, bb0[j], acc[j]);
#pragma unroll
  for (int j = 0; j < 8; ++j) acc[j] = WMMA_BF16(a1, bb1[j], acc[j]);

#pragma unroll
  for (int j = 0; j < 4; ++j) {
    float bl = bias[j * 16 + lj];
    float br = bias[64 + j * 16 + lj];
#pragma unroll
    for (int r = 0; r < 8; ++r) {
      int R = r0 + r + (lh << 3);
      float zl = acc[j][r] + bl;
      float zr = acc[j + 4][r] + br;
      float cur = zl * sigmoidf_(zr);          // z[:, :C] * sigmoid(z[:, C:])
      int c = j * 16 + lj;
      size_t oidx = (size_t)R * 64 + c;
      outmax[oidx] = (layer == 0) ? cur : fmaxf(outmax[oidx], cur);
      if (layer < 2) {
        int n = R >> 5, b = R & 31;            // row = n*32 + b
        curT[(size_t)(b * 64 + c) * CN + n] = (bf16_t)cur;
      }
    }
  }
}

// ---------------- finalize: d_out[b,w,n,c] += outmax[n*32+b, c] ----------------
__global__ void finalize_add(float* __restrict__ out, const float* __restrict__ outmax, int w) {
  int idx = blockIdx.x * blockDim.x + threadIdx.x;    // 32*1024*64
  int b = idx >> 16;
  int rem = idx & 65535;
  int n = rem >> 6, c = rem & 63;
  out[(((size_t)b * CNW + w) * CN + n) * CC + c] += outmax[((size_t)n * 32 + b) * 64 + c];
}

// ---------------- host launch ----------------
extern "C" void kernel_launch(void* const* d_in, const int* in_sizes, int n_in,
                              void* d_out, int out_size, void* d_ws, size_t ws_size,
                              hipStream_t stream) {
  const float* d_data = (const float*)d_in[1];
  const float* d_adj  = (const float*)d_in[2];
  const float* d_temb = (const float*)d_in[3];
  const float* d_semb = (const float*)d_in[4];
  const float* d_w1   = (const float*)d_in[5];
  const float* d_b1   = (const float*)d_in[6];
  const float* d_gw   = (const float*)d_in[7];
  const float* d_gb   = (const float*)d_in[8];
  float* out = (float*)d_out;

  char* base = (char*)d_ws;
  bf16_t* xrT  = (bf16_t*)base;  base += (size_t)CBC * CTN * 2;        // 50.3 MB
  bf16_t* adjb = (bf16_t*)base;  base += (size_t)1024 * 4096 * 2;      //  8.4 MB
  bf16_t* WcT  = (bf16_t*)base;  base += (size_t)128 * 256 * 2;
  bf16_t* gwT  = (bf16_t*)base;  base += (size_t)27 * 8192 * 2;
  float*  Ywing= (float*) base;  base += (size_t)1024 * CBC * 4;       //  8.4 MB
  bf16_t* Ybf  = (bf16_t*)base;  base += (size_t)1024 * CBC * 2;
  bf16_t* curT = (bf16_t*)base;  base += (size_t)CBC * CN * 2;
  float*  omax = (float*) base;  base += (size_t)32768 * 64 * 4;       //  8.4 MB

  prep_xr   <<<32 * 12 * 16, 256, 0, stream>>>(d_data, d_temb, d_semb, xrT);
  prep_adj  <<<16384, 256, 0, stream>>>(d_adj, adjb);
  prep_wconv<<<128,   256, 0, stream>>>(d_w1, WcT);
  prep_gcnw <<<864,   256, 0, stream>>>(d_gw, gwT);

  // data_res -> d_out (full overwrite)
  conv_glu<<<2304, 256, 0, stream>>>(d_data, WcT, d_b1, out);

  for (int w = 0; w < CNW; ++w) {
    // wing partial: adj[:, :3072] @ X_w[:3072]   (K = 3072, f32 out)
    gemm_big<<<64, 256, 0, stream>>>(adjb, 4096, xrT + (size_t)w * CN, CTN,
                                     nullptr, Ywing, nullptr, 3072, 0);
    for (int j = 0; j < 3; ++j) {
      const bf16_t* bt = (j == 0) ? (xrT + (size_t)(w + 3) * CN) : curT;
      int ldb = (j == 0) ? CTN : CN;
      // Y = Ywing + adj[:, 3072:] @ cur   (K = 1024, init, bf16 out)
      gemm_big<<<64, 256, 0, stream>>>(adjb + 3072, 4096, bt, ldb,
                                       Ywing, nullptr, Ybf, 1024, 1);
      // Z = Y @ W_j + b_j ; GLU ; running max ; transposed cur for next layer
      gemm_glu<<<256, 256, 0, stream>>>(Ybf, gwT + (size_t)(w * 3 + j) * 8192,
                                        d_gb + (w * 3 + j) * 128, curT, omax, j);
    }
    finalize_add<<<8192, 256, 0, stream>>>(out, omax, w);
  }
}
